// PaGCNLayer_25443386262267
// MI455X (gfx1250) — matrix operand order repaired and checked by reading, not compiled, not run on hardware
//
#include <hip/hip_runtime.h>
#include <hip/hip_bf16.h>

#define N_NODES 50000
#define IN_F    128
#define OUT_F   128
#define N_EDGES 800000
#define N_TRAIN 5000

typedef __attribute__((ext_vector_type(16))) _Float16 v16h;
typedef __attribute__((ext_vector_type(8)))  _Float16 v8h;
typedef __attribute__((ext_vector_type(8)))  float    v8f;

__device__ __forceinline__ float fast_sigmoid(float v) {
    return 1.0f / (1.0f + __expf(-v));
}

// ---- zero the accumulator region of the workspace ----
__global__ void k_zero(float* __restrict__ p, int n) {
    int i = blockIdx.x * blockDim.x + threadIdx.x;
    if (i < n) p[i] = 0.0f;
}

// ---- scatter train mask: mask[train_id] = 1 ----
__global__ void k_mask(const int* __restrict__ ids, float* __restrict__ mask, int n) {
    int i = blockIdx.x * blockDim.x + threadIdx.x;
    if (i < n) mask[ids[i]] = 1.0f;
}

// ---- edge phase: one wave32 per edge, 4 floats per lane (128 features) ----
// denom[row] += M_eff[col]; Hacc[row] += adj * M_eff[col] * x[col]
__global__ void __launch_bounds__(256)
k_edges(const float* __restrict__ x,   const int* __restrict__ rows,
        const int* __restrict__ cols,  const float* __restrict__ adj,
        const float* __restrict__ M,   const float* __restrict__ mask,
        float* __restrict__ denom,     float* __restrict__ hacc) {
    int wid  = (blockIdx.x * blockDim.x + threadIdx.x) >> 5;  // edge id (grid is exact)
    int lane = threadIdx.x & 31;
    int r  = rows[wid];
    int c  = cols[wid];
    float av = adj[wid];
    float mk = mask[c];                 // >0 => train row => M_eff = 1

    int cb = c * IN_F + lane * 4;
    float4 m4 = *(const float4*)(M + cb);
    float4 x4 = *(const float4*)(x + cb);

    float4 me;
    if (mk > 0.0f) {
        me = make_float4(1.0f, 1.0f, 1.0f, 1.0f);
    } else {
        me.x = fast_sigmoid(m4.x);
        me.y = fast_sigmoid(m4.y);
        me.z = fast_sigmoid(m4.z);
        me.w = fast_sigmoid(m4.w);
    }

    int rb = r * IN_F + lane * 4;
    // accumulators are L2-resident (51 MB < 192 MB L2) -> atomics resolve in L2
    atomicAdd(denom + rb + 0, me.x);
    atomicAdd(denom + rb + 1, me.y);
    atomicAdd(denom + rb + 2, me.z);
    atomicAdd(denom + rb + 3, me.w);
    atomicAdd(hacc  + rb + 0, av * me.x * x4.x);
    atomicAdd(hacc  + rb + 1, av * me.y * x4.y);
    atomicAdd(hacc  + rb + 2, av * me.z * x4.z);
    atomicAdd(hacc  + rb + 3, av * me.w * x4.w);
}

// ---- H = Hacc * (denom==0 ? 0 : 1/denom), converted to f16 for WMMA ----
__global__ void k_hnorm(const float* __restrict__ denom, const float* __restrict__ hacc,
                        _Float16* __restrict__ h16) {
    int i = blockIdx.x * blockDim.x + threadIdx.x;   // exact grid: N_NODES*IN_F
    float d = denom[i];
    float h = (d == 0.0f) ? 0.0f : hacc[i] / d;
    h16[i] = (_Float16)h;
}

// ---- transpose W[k][n] -> Wt[n][k] in f16 (B operand wants column-contiguous K) ----
__global__ void k_wt(const float* __restrict__ W, _Float16* __restrict__ wt) {
    int i = blockIdx.x * blockDim.x + threadIdx.x;   // exact grid: IN_F*OUT_F
    int n = i & (OUT_F - 1);
    int k = i >> 7;
    wt[n * IN_F + k] = (_Float16)W[i];
}

// ---- out = elu(H @ W), one wave32 per 16x16 tile, V_WMMA_F32_16X16X32_F16 ----
__global__ void __launch_bounds__(256)
k_gemm(const _Float16* __restrict__ Hh, const _Float16* __restrict__ Wt,
       float* __restrict__ out) {
    int wid  = (blockIdx.x * blockDim.x + threadIdx.x) >> 5;  // tile id (grid exact)
    int lane = threadIdx.x & 31;
    int tm   = wid >> 3;           // 0..3124  (50000/16 row tiles)
    int tn   = wid & 7;            // 0..7     (128/16 col tiles)
    int half = lane >> 4;
    int l15  = lane & 15;

    const _Float16* hrow = Hh + (size_t)(tm * 16 + l15) * IN_F;  // A: row = lane&15
    const _Float16* wcol = Wt + (size_t)(tn * 16 + l15) * IN_F;  // B: col = lane&15

    v8f acc = {};
#pragma unroll
    for (int kk = 0; kk < IN_F; kk += 32) {
        // A 16x32 f16 layout (ISA 7.12.2): VGPR0-3 = K = kk + 8*half + 0..7,
        //                                  VGPR4-7 = K = kk + 16 + 8*half + 0..7
        v8h a0 = *(const v8h*)(hrow + kk + half * 8);
        v8h a1 = *(const v8h*)(hrow + kk + 16 + half * 8);
        v16h a;
#pragma unroll
        for (int j = 0; j < 8; ++j) { a[j] = a0[j]; a[8 + j] = a1[j]; }
        // B 32x16 f16 layout: same K order as A (kk + 8*half + 0..7, kk + 16 + 8*half + 0..7)
        v8h b0 = *(const v8h*)(wcol + kk + half * 8);
        v8h b1 = *(const v8h*)(wcol + kk + 16 + half * 8);
        v16h b;
#pragma unroll
        for (int j = 0; j < 8; ++j) { b[j] = b0[j]; b[8 + j] = b1[j]; }
        acc = __builtin_amdgcn_wmma_f32_16x16x32_f16(
                  false, a, false, b, (short)0, acc, false, false);
    }

    int ncol = tn * 16 + l15;
#pragma unroll
    for (int v = 0; v < 8; ++v) {
        // C/D layout: VGPR v -> row tm*16 + 8*half + v, column = lane&15
        int row = tm * 16 + half * 8 + v;
        float h = acc[v];
        float e = (h > 0.0f) ? h : (__expf(h) - 1.0f);
        out[(size_t)row * OUT_F + ncol] = e;
    }
}

extern "C" void kernel_launch(void* const* d_in, const int* in_sizes, int n_in,
                              void* d_out, int out_size, void* d_ws, size_t ws_size,
                              hipStream_t stream) {
    const float* x   = (const float*)d_in[0];
    const int*   er  = (const int*)  d_in[1];
    const int*   ec  = (const int*)  d_in[2];
    const float* av  = (const float*)d_in[3];
    const int*   tid = (const int*)  d_in[4];
    const float* M   = (const float*)d_in[5];
    const float* W   = (const float*)d_in[6];
    float* out = (float*)d_out;

    // workspace layout (floats):
    // [denom: N*F][hacc: N*F][mask: N][h16: N*F halves][wt: F*F halves]
    float* ws    = (float*)d_ws;
    float* denom = ws;
    float* hacc  = ws + (size_t)N_NODES * IN_F;
    float* mask  = hacc + (size_t)N_NODES * IN_F;
    _Float16* h16 = (_Float16*)(mask + N_NODES);          // 16B-aligned offset
    _Float16* wt  = h16 + (size_t)N_NODES * IN_F;

    const int zeroN = 2 * N_NODES * IN_F + N_NODES;       // denom + hacc + mask
    k_zero <<<(zeroN + 255) / 256, 256, 0, stream>>>(ws, zeroN);
    k_mask <<<(N_TRAIN + 255) / 256, 256, 0, stream>>>(tid, mask, N_TRAIN);
    k_edges<<<N_EDGES / 8, 256, 0, stream>>>(x, er, ec, av, M, mask, denom, hacc);
    k_hnorm<<<(N_NODES * IN_F) / 256, 256, 0, stream>>>(denom, hacc, h16);
    k_wt   <<<(IN_F * OUT_F) / 256, 256, 0, stream>>>(W, wt);
    k_gemm <<<((N_NODES / 16) * (OUT_F / 16)) / 8, 256, 0, stream>>>(h16, wt, out);
}
